// Block_37374805410454
// MI455X (gfx1250) — matrix operand order repaired and checked by reading, not compiled, not run on hardware
//
#include <hip/hip_runtime.h>
#include <math.h>

#define BATCH 512
#define SEQ   128
#define EMBED 768
#define HEADS 6
#define HDIM  256
#define FF    3072
#define MTOT  (BATCH*SEQ)   /* 65536 rows */

typedef __attribute__((ext_vector_type(16))) __bf16 v16bf;
typedef __attribute__((ext_vector_type(8)))  float  v8f;
typedef __attribute__((ext_vector_type(4)))  unsigned v4u;
typedef __attribute__((ext_vector_type(8)))  int      v8i;
typedef __attribute__((ext_vector_type(4)))  int      v4i;

union ABfrag { v16bf v; uint4 u[2]; };

__device__ __forceinline__ v8f vzero8() {
  v8f z;
#pragma unroll
  for (int i = 0; i < 8; ++i) z[i] = 0.f;
  return z;
}

// Async global -> LDS copy (ASYNCcnt path), 16B per lane.
#define ASYNC_CP_B128(ldsaddr_u32, gptr)                                     \
  asm volatile("global_load_async_to_lds_b128 %0, %1, off"                   \
               :: "v"(ldsaddr_u32),                                          \
                  "v"((unsigned long long)(size_t)(gptr)) : "memory")

#if __has_builtin(__builtin_amdgcn_s_wait_asynccnt)
#define WAIT_ASYNC(n) __builtin_amdgcn_s_wait_asynccnt(n)
#else
#define WAIT_ASYNC(n) asm volatile("s_wait_asynccnt %0" :: "n"(n) : "memory")
#endif

#if __has_builtin(__builtin_amdgcn_s_wait_tensorcnt)
#define WAIT_TENSOR0() __builtin_amdgcn_s_wait_tensorcnt(0)
#else
#define WAIT_TENSOR0() asm volatile("s_wait_tensorcnt 0x0" ::: "memory")
#endif

// ---------------------------------------------------------------------------
// Weight convert + transpose: W [Z][K][N] f32  ->  WT [Z][N][K] bf16
// ---------------------------------------------------------------------------
__global__ __launch_bounds__(256)
void convt_kernel(const float* __restrict__ W, __bf16* __restrict__ WT,
                  int K, int N, long total)
{
  long i = (long)blockIdx.x * 256 + threadIdx.x;
  if (i >= total) return;
  const int  k = (int)(i % K);
  const long t = i / K;
  const int  n = (int)(t % N);
  const int  z = (int)(t / N);
  WT[i] = (__bf16)W[(long)z * K * N + (long)k * N + n];
}

// ---------------------------------------------------------------------------
// LayerNorm over 768 cols, one row per wave (wave32: 24 elems/lane), bf16 out
// ---------------------------------------------------------------------------
__global__ __launch_bounds__(256)
void ln_kernel(const float* __restrict__ in, const float* __restrict__ g,
               const float* __restrict__ be, __bf16* __restrict__ out)
{
  const int wid = threadIdx.x >> 5, lane = threadIdx.x & 31;
  const long row = (long)blockIdx.x * 8 + wid;
  const float* p = in + row * EMBED;
  float v[24];
  float s = 0.f;
#pragma unroll
  for (int i = 0; i < 24; ++i) { v[i] = p[lane + 32*i]; s += v[i]; }
#pragma unroll
  for (int mk = 1; mk < 32; mk <<= 1) s += __shfl_xor(s, mk, 32);
  const float mu = s * (1.f/768.f);
  float vs = 0.f;
#pragma unroll
  for (int i = 0; i < 24; ++i) { float d = v[i] - mu; vs += d*d; }
#pragma unroll
  for (int mk = 1; mk < 32; mk <<= 1) vs += __shfl_xor(vs, mk, 32);
  const float rstd = rsqrtf(vs * (1.f/768.f) + 1e-5f);
  __bf16* po = out + row * EMBED;
#pragma unroll
  for (int i = 0; i < 24; ++i) {
    const int c = lane + 32*i;
    po[c] = (__bf16)((v[i] - mu) * rstd * g[c] + be[c]);
  }
}

// ---------------------------------------------------------------------------
// Generic WMMA bf16 GEMM: C[M,N] = A[M,K] * BT[N,K]^T (+bias)(+res)(relu)
// Tile: 128(M) x 64(N) per 8-wave WG; 32x32 per wave; K-step 32.
// Triple-buffered LDS fed by GLOBAL_LOAD_ASYNC_TO_LDS_B128 (ASYNCcnt),
// one barrier per K-step; copies for tile i+2 overlap WMMA on tile i.
// flags: bit0 = fp32 out, bit1 = relu
// ---------------------------------------------------------------------------
__global__ __launch_bounds__(256)
void gemm_bf16(const __bf16* __restrict__ A, int lda,
               const __bf16* __restrict__ BT, long zstrB,
               const float* __restrict__ bias,
               const float* __restrict__ res, int resStride,
               void* __restrict__ out,
               long zOut, long bOut, long tOut, long nOut,
               int K, int flags)
{
  __shared__ __bf16 sA[3][128 * 40];   // 128 rows, 32 halfs + 8 pad
  __shared__ __bf16 sB[3][64 * 40];    // 64 rows (N),  32 halfs + 8 pad

  const int tid = threadIdx.x;
  const int n0 = blockIdx.x * 64;
  const long m0 = (long)blockIdx.y * 128;
  const int z = blockIdx.z;
  const int wid = tid >> 5, lane = tid & 31, lm = lane & 15, lh = lane >> 4;
  const int mw = (wid >> 1) * 32, nw = (wid & 1) * 32;
  const __bf16* Bz = BT + (long)z * zstrB;

  // per-thread 16B copy chunks: A = 512 chunks (2/thread), B = 256 (1/thread)
  const int crow = tid >> 2, ccol = (tid & 3) * 8;
  const int crow2 = (tid + 256) >> 2;
  const __bf16* gA0 = A + (m0 + crow)  * (long)lda + ccol;
  const __bf16* gA1 = A + (m0 + crow2) * (long)lda + ccol;
  const __bf16* gB  = Bz + (long)(n0 + crow) * K + ccol;

  unsigned aA0[3], aA1[3], aB[3];
#pragma unroll
  for (int p = 0; p < 3; ++p) {
    aA0[p] = (unsigned)(size_t)&sA[p][crow  * 40 + ccol];
    aA1[p] = (unsigned)(size_t)&sA[p][crow2 * 40 + ccol];
    aB[p]  = (unsigned)(size_t)&sB[p][crow  * 40 + ccol];
  }

  auto issue = [&](int p, int k0) {
    ASYNC_CP_B128(aA0[p], gA0 + k0);
    ASYNC_CP_B128(aA1[p], gA1 + k0);
    ASYNC_CP_B128(aB[p],  gB  + k0);
  };

  v8f acc[2][2];
#pragma unroll
  for (int i = 0; i < 2; ++i)
#pragma unroll
    for (int j = 0; j < 2; ++j) acc[i][j] = vzero8();

  const int niter = K >> 5;
  issue(0, 0);
  if (niter > 1) issue(1, 32);

  for (int it = 0, k0 = 0; it < niter; ++it, k0 += 32) {
    if (it + 1 < niter) WAIT_ASYNC(3);   // tile `it` landed (in-order), next in flight
    else                WAIT_ASYNC(0);
    __syncthreads();                      // all waves' tile `it` visible
    if (it + 2 < niter) issue((it + 2) % 3, k0 + 64);  // overlaps compute below

    const __bf16* cA = sA[it % 3];
    const __bf16* cB = sB[it % 3];
    ABfrag fa[2], fb[2];
#pragma unroll
    for (int i = 0; i < 2; ++i) {
      const int mr = mw + i*16 + lm;
      fa[i].u[0] = *(const uint4*)(&cA[mr*40 + 8*lh]);
      fa[i].u[1] = *(const uint4*)(&cA[mr*40 + 16 + 8*lh]);
      const int nr = nw + i*16 + lm;
      fb[i].u[0] = *(const uint4*)(&cB[nr*40 + 8*lh]);
      fb[i].u[1] = *(const uint4*)(&cB[nr*40 + 16 + 8*lh]);
    }
#pragma unroll
    for (int i = 0; i < 2; ++i)
#pragma unroll
      for (int j = 0; j < 2; ++j)
        acc[i][j] = __builtin_amdgcn_wmma_f32_16x16x32_bf16(
            false, fa[i].v, false, fb[j].v, (short)0, acc[i][j], false, false);
  }

  const long bb = blockIdx.y;
#pragma unroll
  for (int i = 0; i < 2; ++i) {
#pragma unroll
    for (int j = 0; j < 2; ++j) {
      const int n = n0 + nw + j*16 + lm;
      const float bv = bias ? bias[n] : 0.f;
#pragma unroll
      for (int r = 0; r < 8; ++r) {
        const int t = mw + i*16 + r + 8*lh;
        float v = acc[i][j][r] + bv;
        if (res) v += res[(m0 + t) * (long)resStride + n];
        if (flags & 2) v = fmaxf(v, 0.f);
        const long oidx = (long)z*zOut + bb*bOut + (long)t*tOut + (long)n*nOut;
        if (flags & 1) ((float*)out)[oidx] = v;
        else           ((__bf16*)out)[oidx] = (__bf16)v;
      }
    }
  }
}

// ---------------------------------------------------------------------------
// Fused causal attention per (b,h): S = Q K^T * scale, mask, softmax, O = P V
// K (128x256) and V^T (256x128) tiles land in padded LDS via the Tensor Data
// Mover (TENSOR_LOAD_TO_LDS + s_wait_tensorcnt); Q A-frags stream from global.
// Writes O directly into cat layout [B,T,H*D] (bf16).
// ---------------------------------------------------------------------------
__global__ __launch_bounds__(256)
void attn_kernel(const __bf16* __restrict__ q,
                 const __bf16* __restrict__ k,
                 const __bf16* __restrict__ vT,
                 __bf16* __restrict__ cat)
{
  extern __shared__ char smem[];
  __bf16* Ks = (__bf16*)smem;                               // [128][264]
  __bf16* Ps = (__bf16*)(smem + 128*264*2);                 // [128][136]
  __bf16* Vs = (__bf16*)(smem + 128*264*2 + 128*136*2);     // [256][136]

  const int bh = blockIdx.x;
  const int b = bh / HEADS, h = bh % HEADS;
  const __bf16* qg = q  + (long)bh * SEQ * HDIM;
  const __bf16* kg = k  + (long)bh * SEQ * HDIM;
  const __bf16* vg = vT + (long)bh * HDIM * SEQ;
  const int tid = threadIdx.x;

#if __has_builtin(__builtin_amdgcn_tensor_load_to_lds)
  if (tid < 32) {   // one wave issues both TDM descriptors
    // ---- K tile: 2D 128 rows x 256 elems, bf16, pad 4 DWORDs per 512B row
    const unsigned long long kga = (unsigned long long)(size_t)kg;
    const v4u g0k = { 1u,                              // count=1
                      (unsigned)(size_t)Ks,            // lds_addr
                      (unsigned)kga,                   // global_addr lo
                      (unsigned)(kga >> 32) | (2u << 30) };  // addr hi | type=2
    const v8i g1k = { (int)((1u<<16) | (1u<<20) | (6u<<22) | (3u<<25)),
                      //  data_size=2B | pad_en | pad_int=128DW | pad_amt=4DW
                      (int)(256u << 16),               // tensor_dim0 = 256
                      (int)(128u << 16),               // tensor_dim1 = 128
                      (int)(256u << 16),               // tile_dim0   = 256
                      128,                             // tile_dim1=128, tile_dim2=0
                      256,                             // tensor_dim0_stride = 256
                      0, 0 };
    // ---- V^T tile: 256 rows x 128 elems, pad 4 DWORDs per 256B row
    const unsigned long long vga = (unsigned long long)(size_t)vg;
    const v4u g0v = { 1u, (unsigned)(size_t)Vs,
                      (unsigned)vga, (unsigned)(vga >> 32) | (2u << 30) };
    const v8i g1v = { (int)((1u<<16) | (1u<<20) | (5u<<22) | (3u<<25)),
                      (int)(128u << 16),               // tensor_dim0 = 128
                      (int)(256u << 16),               // tensor_dim1 = 256
                      (int)(128u << 16),               // tile_dim0   = 128
                      256,                             // tile_dim1 = 256
                      128,                             // tensor_dim0_stride
                      0, 0 };
    const v4i z4 = {0, 0, 0, 0};
#if defined(__clang_major__) && __clang_major__ >= 23
    const v8i z8 = {0,0,0,0,0,0,0,0};
    __builtin_amdgcn_tensor_load_to_lds(g0k, g1k, z4, z4, z8, 0);
    __builtin_amdgcn_tensor_load_to_lds(g0v, g1v, z4, z4, z8, 0);
#else
    __builtin_amdgcn_tensor_load_to_lds(g0k, g1k, z4, z4, 0);
    __builtin_amdgcn_tensor_load_to_lds(g0v, g1v, z4, z4, 0);
#endif
    WAIT_TENSOR0();
  }
#else
  // fallback: manual cooperative staging
#pragma unroll
  for (int i = 0; i < 16; ++i) {            // K: 128x256 halfs
    const int c = tid + 256*i;
    const int row = c >> 5, c8 = c & 31;
    *(uint4*)(&Ks[row*264 + c8*8]) = *(const uint4*)(kg + row*HDIM + c8*8);
  }
#pragma unroll
  for (int i = 0; i < 16; ++i) {            // V^T: 256x128 halfs
    const int c = tid + 256*i;
    const int row = c >> 4, c8 = c & 15;
    *(uint4*)(&Vs[row*136 + c8*8]) = *(const uint4*)(vg + row*SEQ + c8*8);
  }
#endif
  __syncthreads();

  const int wid = tid >> 5, lane = tid & 31, lm = lane & 15, lh = lane >> 4;
  const int m0 = wid * 16;                  // each wave owns 16 query rows

  // ---- phase 1: S = Q K^T -------------------------------------------------
  v8f S[8];
#pragma unroll
  for (int j = 0; j < 8; ++j) S[j] = vzero8();

  for (int kk = 0; kk < HDIM; kk += 32) {
    ABfrag fa;
    fa.u[0] = *(const uint4*)(qg + (m0 + lm)*HDIM + kk + 8*lh);
    fa.u[1] = *(const uint4*)(qg + (m0 + lm)*HDIM + kk + 16 + 8*lh);
#pragma unroll
    for (int j = 0; j < 8; ++j) {
      ABfrag fb;
      fb.u[0] = *(const uint4*)(&Ks[(j*16 + lm)*264 + kk + 8*lh]);
      fb.u[1] = *(const uint4*)(&Ks[(j*16 + lm)*264 + kk + 16 + 8*lh]);
      S[j] = __builtin_amdgcn_wmma_f32_16x16x32_bf16(
          false, fa.v, false, fb.v, (short)0, S[j], false, false);
    }
  }

  // ---- causal mask + softmax (rows live per-lane-half, cols across lanes) --
  const float scale = 0.03608439182435161f;   // 768^-0.5 (embed, per reference)
#pragma unroll
  for (int r = 0; r < 8; ++r) {
    const int row = m0 + r + 8*lh;
    float mx = -INFINITY;
#pragma unroll
    for (int j = 0; j < 8; ++j) {
      const int col = j*16 + lm;
      const float v = (col <= row) ? S[j][r] * scale : -INFINITY;
      S[j][r] = v;
      mx = fmaxf(mx, v);
    }
#pragma unroll
    for (int mk = 1; mk < 16; mk <<= 1) mx = fmaxf(mx, __shfl_xor(mx, mk, 32));
    float sum = 0.f;
#pragma unroll
    for (int j = 0; j < 8; ++j) {
      const float e = __expf(S[j][r] - mx);
      S[j][r] = e;
      sum += e;
    }
#pragma unroll
    for (int mk = 1; mk < 16; mk <<= 1) sum += __shfl_xor(sum, mk, 32);
    const float rs = 1.f / sum;
#pragma unroll
    for (int j = 0; j < 8; ++j)
      Ps[row*136 + j*16 + lm] = (__bf16)(S[j][r] * rs);
  }
  // P rows are wave-private: no cross-wave barrier needed before phase 2.

  // ---- phase 2: O = P V, streamed to cat [B,T,H*D] ------------------------
  const long catbase = (long)b * SEQ * (HEADS*HDIM) + (long)h * HDIM;
#pragma unroll
  for (int nt = 0; nt < 4; ++nt) {          // 4 x 64 output cols
    v8f O[4];
#pragma unroll
    for (int c4 = 0; c4 < 4; ++c4) O[c4] = vzero8();
    for (int ss = 0; ss < SEQ; ss += 32) {
      ABfrag fa;
      fa.u[0] = *(const uint4*)(&Ps[(m0 + lm)*136 + ss + 8*lh]);
      fa.u[1] = *(const uint4*)(&Ps[(m0 + lm)*136 + ss + 16 + 8*lh]);
#pragma unroll
      for (int c4 = 0; c4 < 4; ++c4) {
        const int n = nt*64 + c4*16;
        ABfrag fb;
        fb.u[0] = *(const uint4*)(&Vs[(n + lm)*136 + ss + 8*lh]);
        fb.u[1] = *(const uint4*)(&Vs[(n + lm)*136 + ss + 16 + 8*lh]);
        O[c4] = __builtin_amdgcn_wmma_f32_16x16x32_bf16(
            false, fa.v, false, fb.v, (short)0, O[c4], false, false);
      }
    }
#pragma unroll
    for (int c4 = 0; c4 < 4; ++c4)
#pragma unroll
      for (int r = 0; r < 8; ++r) {
        const int t = m0 + r + 8*lh;
        const int col = nt*64 + c4*16 + lm;
        cat[catbase + (long)t * (HEADS*HDIM) + col] = (__bf16)O[c4][r];
      }
  }
}

// ---------------------------------------------------------------------------
extern "C" void kernel_launch(void* const* d_in, const int* in_sizes, int n_in,
                              void* d_out, int out_size, void* d_ws, size_t ws_size,
                              hipStream_t stream)
{
  (void)in_sizes; (void)n_in; (void)out_size; (void)ws_size;

  const float* x   = (const float*)d_in[0];
  const float* Wq  = (const float*)d_in[1];
  const float* Wk  = (const float*)d_in[2];
  const float* Wv  = (const float*)d_in[3];
  const float* Wo  = (const float*)d_in[4];
  const float* bo  = (const float*)d_in[5];
  const float* W1  = (const float*)d_in[6];
  const float* b1  = (const float*)d_in[7];
  const float* W2  = (const float*)d_in[8];
  const float* b2  = (const float*)d_in[9];
  const float* g1  = (const float*)d_in[10];
  const float* be1 = (const float*)d_in[11];
  const float* g2  = (const float*)d_in[12];
  const float* be2 = (const float*)d_in[13];

  char* ws = (char*)d_ws;
  size_t off = 0;
  auto carve = [&](size_t bytes) -> char* {
    char* p = ws + off;
    off += (bytes + 255) & ~(size_t)255;
    return p;
  };

  __bf16* WqT = (__bf16*)carve((size_t)HEADS*HDIM*EMBED*2);   // [H][D][E]
  __bf16* WkT = (__bf16*)carve((size_t)HEADS*HDIM*EMBED*2);
  __bf16* WvT = (__bf16*)carve((size_t)HEADS*HDIM*EMBED*2);
  __bf16* WoT = (__bf16*)carve((size_t)EMBED*HEADS*HDIM*2);   // [E][H*D]
  __bf16* W1T = (__bf16*)carve((size_t)FF*EMBED*2);           // [FF][E]
  __bf16* W2T = (__bf16*)carve((size_t)EMBED*FF*2);           // [E][FF]
  __bf16* hln = (__bf16*)carve((size_t)MTOT*EMBED*2);
  __bf16* qb  = (__bf16*)carve((size_t)BATCH*HEADS*SEQ*HDIM*2);
  __bf16* kb  = (__bf16*)carve((size_t)BATCH*HEADS*SEQ*HDIM*2);
  __bf16* vTb = (__bf16*)carve((size_t)BATCH*HEADS*HDIM*SEQ*2);
  __bf16* catb= (__bf16*)carve((size_t)MTOT*HEADS*HDIM*2);
  float*  aout= (float*) carve((size_t)MTOT*EMBED*4);
  __bf16* h2  = (__bf16*)carve((size_t)MTOT*EMBED*2);
  __bf16* t1  = (__bf16*)carve((size_t)MTOT*FF*2);

  const dim3 blk(256);

  // -- weights -> bf16, transposed to [N][K]
  convt_kernel<<<dim3((HEADS*EMBED*HDIM)/256), blk, 0, stream>>>(Wq, WqT, EMBED, HDIM, (long)HEADS*EMBED*HDIM);
  convt_kernel<<<dim3((HEADS*EMBED*HDIM)/256), blk, 0, stream>>>(Wk, WkT, EMBED, HDIM, (long)HEADS*EMBED*HDIM);
  convt_kernel<<<dim3((HEADS*EMBED*HDIM)/256), blk, 0, stream>>>(Wv, WvT, EMBED, HDIM, (long)HEADS*EMBED*HDIM);
  convt_kernel<<<dim3((HEADS*HDIM*EMBED)/256), blk, 0, stream>>>(Wo, WoT, HEADS*HDIM, EMBED, (long)HEADS*HDIM*EMBED);
  convt_kernel<<<dim3((EMBED*FF)/256),         blk, 0, stream>>>(W1, W1T, EMBED, FF, (long)EMBED*FF);
  convt_kernel<<<dim3((FF*EMBED)/256),         blk, 0, stream>>>(W2, W2T, FF, EMBED, (long)FF*EMBED);

  // -- LN1
  ln_kernel<<<dim3(MTOT/8), blk, 0, stream>>>(x, g1, be1, hln);

  // -- QKV projections (grid.z = heads). q/k: [B,H,T,D]; v: transposed [B,H,D,T]
  gemm_bf16<<<dim3(HDIM/64, MTOT/128, HEADS), blk, 0, stream>>>(
      hln, EMBED, WqT, (long)HDIM*EMBED, nullptr, nullptr, 0,
      qb, (long)SEQ*HDIM, (long)HEADS*SEQ*HDIM, HDIM, 1, EMBED, 0);
  gemm_bf16<<<dim3(HDIM/64, MTOT/128, HEADS), blk, 0, stream>>>(
      hln, EMBED, WkT, (long)HDIM*EMBED, nullptr, nullptr, 0,
      kb, (long)SEQ*HDIM, (long)HEADS*SEQ*HDIM, HDIM, 1, EMBED, 0);
  gemm_bf16<<<dim3(HDIM/64, MTOT/128, HEADS), blk, 0, stream>>>(
      hln, EMBED, WvT, (long)HDIM*EMBED, nullptr, nullptr, 0,
      vTb, (long)HDIM*SEQ, (long)HEADS*HDIM*SEQ, 1, SEQ, EMBED, 0);

  // -- fused causal attention -> cat [B,T,H*D]
  const size_t attn_lds = (size_t)128*264*2 + (size_t)128*136*2 + (size_t)256*136*2; // 172032 B
  attn_kernel<<<dim3(BATCH*HEADS), blk, attn_lds, stream>>>(qb, kb, vTb, catb);

  // -- output projection + bias + residual x -> attn_out (fp32)
  gemm_bf16<<<dim3(EMBED/64, MTOT/128, 1), blk, 0, stream>>>(
      catb, HEADS*HDIM, WoT, 0, bo, x, EMBED,
      aout, 0, (long)SEQ*EMBED, EMBED, 1, HEADS*HDIM, 1);

  // -- LN2
  ln_kernel<<<dim3(MTOT/8), blk, 0, stream>>>(aout, g2, be2, h2);

  // -- MLP up + ReLU (bf16)
  gemm_bf16<<<dim3(FF/64, MTOT/128, 1), blk, 0, stream>>>(
      h2, EMBED, W1T, 0, b1, nullptr, 0,
      t1, 0, (long)SEQ*FF, FF, 1, EMBED, 2);

  // -- MLP down + bias + residual attn_out -> d_out (fp32)
  gemm_bf16<<<dim3(EMBED/64, MTOT/128, 1), blk, 0, stream>>>(
      t1, FF, W2T, 0, b2, aout, EMBED,
      d_out, 0, (long)SEQ*EMBED, EMBED, 1, FF, 1);
}
